// VNN_ResnetPointnet_21603685499638
// MI455X (gfx1250) — matrix operand order repaired and checked by reading, not compiled, not run on hardware
//
#include <hip/hip_runtime.h>

// ---------------------------------------------------------------------------
// Types for WMMA (gfx1250, wave32)
// ---------------------------------------------------------------------------
typedef __attribute__((ext_vector_type(16))) __bf16        bf16x16;
typedef __attribute__((ext_vector_type(8)))  float         f32x8;
typedef __attribute__((ext_vector_type(4)))  unsigned int  u32x4;

union FragU { bf16x16 v; u32x4 q[2]; };

__device__ __forceinline__ unsigned short f2bf(float f) {
  union { float f; unsigned int u; } x; x.f = f;
  unsigned int r = x.u + 0x7FFFu + ((x.u >> 16) & 1u);   // round-to-nearest-even
  return (unsigned short)(r >> 16);
}
__device__ __forceinline__ float bf2f(unsigned short h) {
  union { unsigned int u; float f; } x; x.u = ((unsigned int)h) << 16;
  return x.f;
}

#define NPTS 4096
#define BB   2
#define KNN  20
#define MTOK (BB * 3 * NPTS)   // 24576 tokens

// ---------------------------------------------------------------------------
// f32 -> bf16 convert (weights)
// ---------------------------------------------------------------------------
__global__ void f32_to_bf16_kernel(const float* __restrict__ src,
                                   unsigned short* __restrict__ dst, int n) {
  int i = blockIdx.x * blockDim.x + threadIdx.x;
  if (i < n) dst[i] = f2bf(src[i]);
}

// ---------------------------------------------------------------------------
// p [B,3,N] * SCALE -> xt [B,N,3]
// ---------------------------------------------------------------------------
__global__ void scale_transpose_kernel(const float* __restrict__ p,
                                       float* __restrict__ xt) {
  int tid = blockIdx.x * blockDim.x + threadIdx.x;
  int total = BB * 3 * NPTS;
  if (tid >= total) return;
  int n = tid % NPTS; int rem = tid / NPTS;
  int a = rem % 3;    int b = rem / 3;
  xt[((size_t)b * NPTS + n) * 3 + a] = p[tid] * 10.0f;
}

// ---------------------------------------------------------------------------
// KNN: per-point top-20 by pdist = 2<xi,xj> - |xi|^2 - |xj|^2 (largest = nearest)
// one thread per point, LDS-tiled candidate scan, register insertion list.
// ---------------------------------------------------------------------------
__global__ __launch_bounds__(128)
void knn_kernel(const float* __restrict__ xt, int* __restrict__ idx) {
  const int TILE = 256;
  __shared__ float sx[TILE], sy[TILE], sz[TILE], ss[TILE];
  int b = blockIdx.x >> 5;                    // 32 blocks of 128 threads per batch
  int n = ((blockIdx.x & 31) << 7) + threadIdx.x;
  const float* base = xt + (size_t)b * NPTS * 3;
  float px = base[n * 3 + 0], py = base[n * 3 + 1], pz = base[n * 3 + 2];
  float ps = px * px + py * py + pz * pz;
  float bv[KNN]; int bi[KNN];
#pragma unroll
  for (int s = 0; s < KNN; ++s) { bv[s] = -3.4e38f; bi[s] = 0; }
  for (int t0 = 0; t0 < NPTS; t0 += TILE) {
    __syncthreads();
    for (int j = threadIdx.x; j < TILE; j += 128) {
      float x = base[(t0 + j) * 3 + 0];
      float y = base[(t0 + j) * 3 + 1];
      float z = base[(t0 + j) * 3 + 2];
      sx[j] = x; sy[j] = y; sz[j] = z; ss[j] = x * x + y * y + z * z;
    }
    __syncthreads();
    for (int j = 0; j < TILE; ++j) {
      float v = 2.0f * (px * sx[j] + py * sy[j] + pz * sz[j]) - ps - ss[j];
      if (v > bv[KNN - 1]) {
        bv[KNN - 1] = v; bi[KNN - 1] = t0 + j;
#pragma unroll
        for (int s = KNN - 1; s > 0; --s) {
          if (bv[s] > bv[s - 1]) {
            float tv = bv[s]; bv[s] = bv[s - 1]; bv[s - 1] = tv;
            int   ti = bi[s]; bi[s] = bi[s - 1]; bi[s - 1] = ti;
          }
        }
      }
    }
  }
  int* op = idx + ((size_t)b * NPTS + n) * KNN;
#pragma unroll
  for (int s = 0; s < KNN; ++s) op[s] = bi[s];
}

// ---------------------------------------------------------------------------
// Edge features + conv_pos (3->128) + VN-leaky(0.2) + mean over k.
// One block per (b,n), thread o = output channel. Writes act0 bf16,
// token-major layout: act0[((b*3+v)*N + n)*128 + o].
// ---------------------------------------------------------------------------
__global__ __launch_bounds__(128)
void conv_pos_kernel(const float* __restrict__ xt, const int* __restrict__ idx,
                     const float* __restrict__ w,  const float* __restrict__ wd,
                     unsigned short* __restrict__ act0) {
  __shared__ float nbx[KNN], nby[KNN], nbz[KNN];
  __shared__ float ctr[3];
  int b = blockIdx.x / NPTS;
  int n = blockIdx.x % NPTS;
  const float* base = xt + (size_t)b * NPTS * 3;
  int t = threadIdx.x;
  if (t < KNN) {
    int j = idx[((size_t)b * NPTS + n) * KNN + t];
    nbx[t] = base[j * 3 + 0]; nby[t] = base[j * 3 + 1]; nbz[t] = base[j * 3 + 2];
  } else if (t < KNN + 3) {
    ctr[t - KNN] = base[n * 3 + (t - KNN)];
  }
  __syncthreads();
  int o = t;
  float w0 = w[o * 3 + 0],  w1 = w[o * 3 + 1],  w2 = w[o * 3 + 2];
  float e0 = wd[o * 3 + 0], e1 = wd[o * 3 + 1], e2 = wd[o * 3 + 2];
  float cx = ctr[0], cy = ctr[1], cz = ctr[2];
  float a0 = 0.f, a1 = 0.f, a2 = 0.f;
  for (int k = 0; k < KNN; ++k) {
    float nx = nbx[k], ny = nby[k], nz = nbz[k];
    float f0x = nx - cx, f0y = ny - cy, f0z = nz - cz;
    float f2x = ny * cz - nz * cy;
    float f2y = nz * cx - nx * cz;
    float f2z = nx * cy - ny * cx;
    float qx = w0 * f0x + w1 * cx + w2 * f2x;
    float qy = w0 * f0y + w1 * cy + w2 * f2y;
    float qz = w0 * f0z + w1 * cz + w2 * f2z;
    float dx = e0 * f0x + e1 * cx + e2 * f2x;
    float dy = e0 * f0y + e1 * cy + e2 * f2y;
    float dz = e0 * f0z + e1 * cz + e2 * f2z;
    float dot = qx * dx + qy * dy + qz * dz;
    float d2  = dx * dx + dy * dy + dz * dz + 1e-6f;
    float s = dot / d2;
    float ox = 0.2f * qx + 0.8f * ((dot >= 0.f) ? qx : (qx - s * dx));
    float oy = 0.2f * qy + 0.8f * ((dot >= 0.f) ? qy : (qy - s * dy));
    float oz = 0.2f * qz + 0.8f * ((dot >= 0.f) ? qz : (qz - s * dz));
    a0 += ox; a1 += oy; a2 += oz;
  }
  const float inv = 1.0f / (float)KNN;
  act0[((size_t)((b * 3 + 0) * NPTS + n)) * 128 + o] = f2bf(a0 * inv);
  act0[((size_t)((b * 3 + 1) * NPTS + n)) * 128 + o] = f2bf(a1 * inv);
  act0[((size_t)((b * 3 + 2) * NPTS + n)) * 128 + o] = f2bf(a2 * inv);
}

// ---------------------------------------------------------------------------
// WMMA GEMM: C[M x N] = A[M x K] * W^T, A/W bf16 row-major (W is [out][in]).
// Block = 8 waves; the block's 64-row weight tile is staged once into LDS
// (row-padded by 8 elems so B-fragment ds_load_b128 is bank-conflict-free).
// Each wave computes a 16x64 tile; K is compile-time -> fully unrolled k-loop.
// MODE 0: store f32; MODE 1: accumulate into f32; MODE 2: store bf16.
// Requires M%128==0, N%64==0, K%32==0.
// ---------------------------------------------------------------------------
template <int MODE, int K>
__global__ __launch_bounds__(256)
void gemm_wmma_kernel(const unsigned short* __restrict__ A,
                      const unsigned short* __restrict__ W,
                      void* __restrict__ Cout, int N) {
  const int KP = K + 8;                     // padded LDS row stride (elements)
  __shared__ unsigned short wlds[64 * KP];

  int nb = blockIdx.y * 64;
  // cooperative stage of W[nb..nb+63][0..K) into LDS (16B chunks)
  for (int i = threadIdx.x; i < 64 * (K / 8); i += 256) {
    int r  = i / (K / 8);
    int kc = (i % (K / 8)) * 8;
    u32x4 vv = *(const u32x4*)(W + (size_t)(nb + r) * K + kc);
    *(u32x4*)(wlds + r * KP + kc) = vv;
  }
  __syncthreads();

  int wave = threadIdx.x >> 5;
  int lane = threadIdx.x & 31;
  int tileM = blockIdx.x * 128 + wave * 16;
  int lm = lane & 15;
  int hv = lane >> 4;
  f32x8 acc[4];
#pragma unroll
  for (int t = 0; t < 4; ++t)
#pragma unroll
    for (int e = 0; e < 8; ++e) acc[t][e] = 0.0f;

  const unsigned short* arow = A + (size_t)(tileM + lm) * K;
#pragma unroll
  for (int k0 = 0; k0 < K; k0 += 32) {
    FragU a;
    a.q[0] = *(const u32x4*)(arow + k0 + hv * 8);        // K k0+hv*8   .. +7
    a.q[1] = *(const u32x4*)(arow + k0 + 16 + hv * 8);   // K k0+16+hv*8.. +7
#pragma unroll
    for (int t = 0; t < 4; ++t) {
      const unsigned short* brow = wlds + (t * 16 + lm) * KP + k0 + hv * 8;
      FragU bfr;
      bfr.q[0] = *(const u32x4*)(brow);
      bfr.q[1] = *(const u32x4*)(brow + 16);
      acc[t] = __builtin_amdgcn_wmma_f32_16x16x32_bf16(
          false, a.v, false, bfr.v, (short)0, acc[t], false, false);
    }
  }
  int col0 = nb + lm;
  int row0 = tileM + hv * 8;
#pragma unroll
  for (int t = 0; t < 4; ++t) {
#pragma unroll
    for (int r = 0; r < 8; ++r) {
      size_t o = (size_t)(row0 + r) * N + (col0 + t * 16);
      float vv = acc[t][r];
      if (MODE == 0)      ((float*)Cout)[o] = vv;
      else if (MODE == 1) ((float*)Cout)[o] += vv;
      else                ((unsigned short*)Cout)[o] = f2bf(vv);
    }
  }
}

// ---------------------------------------------------------------------------
// VN leaky: q,d bf16 token-major [(b*3+v)*N+n][C]; gathers v=0..2, writes bf16.
// ---------------------------------------------------------------------------
__global__ void vn_leaky_kernel(const unsigned short* __restrict__ Q,
                                const unsigned short* __restrict__ D,
                                unsigned short* __restrict__ O,
                                int C, float slope, int total) {
  int tid = blockIdx.x * blockDim.x + threadIdx.x;
  if (tid >= total) return;
  int c = tid % C; int rem = tid / C;
  int n = rem % NPTS; int b = rem / NPTS;
  float q[3], d[3];
#pragma unroll
  for (int v = 0; v < 3; ++v) {
    size_t a = ((size_t)((b * 3 + v) * NPTS + n)) * C + c;
    q[v] = bf2f(Q[a]); d[v] = bf2f(D[a]);
  }
  float dot = q[0] * d[0] + q[1] * d[1] + q[2] * d[2];
  float d2  = d[0] * d[0] + d[1] * d[1] + d[2] * d[2] + 1e-6f;
  float s = dot / d2;
#pragma unroll
  for (int v = 0; v < 3; ++v) {
    float sel = (dot >= 0.f) ? q[v] : (q[v] - s * d[v]);
    float o = slope * q[v] + (1.f - slope) * sel;
    O[((size_t)((b * 3 + v) * NPTS + n)) * C + c] = f2bf(o);
  }
}

// ---------------------------------------------------------------------------
// Mean over n: src [(g)*N + n][C] f32 -> dst [g*C + c] f32, g = b*3+v.
// ---------------------------------------------------------------------------
__global__ void pool_mean_kernel(const float* __restrict__ src,
                                 float* __restrict__ dst, int C) {
  int tid = blockIdx.x * blockDim.x + threadIdx.x;
  int total = BB * 3 * C;
  if (tid >= total) return;
  int c = tid % C; int g = tid / C;
  const float* p = src + (size_t)g * NPTS * C + c;
  float s = 0.f;
  for (int n = 0; n < NPTS; ++n) s += p[(size_t)n * C];
  dst[tid] = s * (1.0f / (float)NPTS);
}

// ---------------------------------------------------------------------------
// Build next block input: out[t][0..127]=bf16(net f32), [128..255]=bf16(pool).
// ---------------------------------------------------------------------------
__global__ void concat_kernel(const float* __restrict__ net,
                              const float* __restrict__ pool,
                              unsigned short* __restrict__ out) {
  size_t tid = blockIdx.x * (size_t)blockDim.x + threadIdx.x;
  size_t total = (size_t)MTOK * 256;
  if (tid >= total) return;
  int c = (int)(tid & 255);
  size_t t = tid >> 8;
  int g = (int)(t / NPTS);
  float v = (c < 128) ? net[t * 128 + c] : pool[g * 128 + (c - 128)];
  out[tid] = f2bf(v);
}

// ---------------------------------------------------------------------------
// Final VAE head: one block, 128 threads (thread = channel), loops b=0..1.
// netm layout [g=b*3+v][128] f32. Output: z_so3[768], z_inv[256], kl[1].
// ---------------------------------------------------------------------------
__global__ __launch_bounds__(128)
void final_head_kernel(const float* __restrict__ netm,
                       const float* __restrict__ actvn_c_d,
                       const float* __restrict__ fc_c_w,
                       const float* __restrict__ fc_inv_w,
                       const float* __restrict__ fc_mu_w,
                       const float* __restrict__ fc_var_w,
                       const float* __restrict__ fc_var_b,
                       const float* __restrict__ fc_inv_2_w,
                       const float* __restrict__ eps,
                       float* __restrict__ out) {
  __shared__ float sm[3][128];
  __shared__ float ci[128];
  __shared__ float red[128];
  __shared__ float klred[128];
  int c = threadIdx.x;
  klred[c] = 0.f;
  const float SQC = 11.3137084989847603904f;   // sqrt(128)
  const float LOGK = -1.38629436111989062f;    // log(0.25)
  for (int b = 0; b < BB; ++b) {
    __syncthreads();
    float nm[3];
#pragma unroll
    for (int v = 0; v < 3; ++v) { nm[v] = netm[(b * 3 + v) * 128 + c]; sm[v][c] = nm[v]; }
    __syncthreads();
    // d = actvn_c_d . netm
    float dd[3] = {0.f, 0.f, 0.f};
    for (int c2 = 0; c2 < 128; ++c2) {
      float w = actvn_c_d[c * 128 + c2];
      dd[0] += w * sm[0][c2]; dd[1] += w * sm[1][c2]; dd[2] += w * sm[2][c2];
    }
    float dot = nm[0] * dd[0] + nm[1] * dd[1] + nm[2] * dd[2];
    float d2  = dd[0] * dd[0] + dd[1] * dd[1] + dd[2] * dd[2] + 1e-6f;
    float s = dot / d2;
    float h[3];
#pragma unroll
    for (int v = 0; v < 3; ++v) {
      float sel = (dot >= 0.f) ? nm[v] : (nm[v] - s * dd[v]);
      h[v] = 0.2f * nm[v] + 0.8f * sel;
    }
    __syncthreads();
#pragma unroll
    for (int v = 0; v < 3; ++v) sm[v][c] = h[v];
    __syncthreads();
    // cv = fc_c_w . h
    float cv[3] = {0.f, 0.f, 0.f};
    for (int c2 = 0; c2 < 128; ++c2) {
      float w = fc_c_w[c * 128 + c2];
      cv[0] += w * sm[0][c2]; cv[1] += w * sm[1][c2]; cv[2] += w * sm[2][c2];
    }
    // cevn(cv)
    float nrm = sqrtf(cv[0] * cv[0] + cv[1] * cv[1] + cv[2] * cv[2]);
    red[c] = nrm * nrm;
    __syncthreads();
    float tot = 0.f;
    for (int i = 0; i < 128; ++i) tot += red[i];
    float chn = sqrtf(tot);
    float f = (1.f / fmaxf(nrm, 1e-12f)) * (nrm / fmaxf(chn, 1e-12f)) * SQC;
#pragma unroll
    for (int v = 0; v < 3; ++v) cv[v] *= f;
    __syncthreads();
#pragma unroll
    for (int v = 0; v < 3; ++v) sm[v][c] = cv[v];   // sm = c
    __syncthreads();
    // t1 = cevn(fc_inv_w . c); c_inv = sum_v t1*c
    float t1[3] = {0.f, 0.f, 0.f};
    for (int c2 = 0; c2 < 128; ++c2) {
      float w = fc_inv_w[c * 128 + c2];
      t1[0] += w * sm[0][c2]; t1[1] += w * sm[1][c2]; t1[2] += w * sm[2][c2];
    }
    float nrm1 = sqrtf(t1[0] * t1[0] + t1[1] * t1[1] + t1[2] * t1[2]);
    red[c] = nrm1 * nrm1;
    __syncthreads();
    float tot1 = 0.f;
    for (int i = 0; i < 128; ++i) tot1 += red[i];
    float chn1 = sqrtf(tot1);
    float f1 = (1.f / fmaxf(nrm1, 1e-12f)) * (nrm1 / fmaxf(chn1, 1e-12f)) * SQC;
    float cinv = (t1[0] * f1) * cv[0] + (t1[1] * f1) * cv[1] + (t1[2] * f1) * cv[2];
    ci[c] = cinv;
    // mu = fc_mu_w . c
    float mu[3] = {0.f, 0.f, 0.f};
    for (int c2 = 0; c2 < 128; ++c2) {
      float w = fc_mu_w[c * 128 + c2];
      mu[0] += w * sm[0][c2]; mu[1] += w * sm[1][c2]; mu[2] += w * sm[2][c2];
    }
    __syncthreads();
    // log_var = fc_var_w . c_inv + b
    float lv = fc_var_b[c];
    for (int c2 = 0; c2 < 128; ++c2) lv += fc_var_w[c * 128 + c2] * ci[c2];
    float sd = expf(0.5f * lv);
    float z[3];
#pragma unroll
    for (int v = 0; v < 3; ++v) {
      z[v] = eps[((size_t)b * 128 + c) * 3 + v] * sd + mu[v];
      out[((size_t)b * 128 + c) * 3 + v] = z[v];
    }
    __syncthreads();
#pragma unroll
    for (int v = 0; v < 3; ++v) sm[v][c] = z[v];
    __syncthreads();
    float t2[3] = {0.f, 0.f, 0.f};
    for (int c2 = 0; c2 < 128; ++c2) {
      float w = fc_inv_2_w[c * 128 + c2];
      t2[0] += w * sm[0][c2]; t2[1] += w * sm[1][c2]; t2[2] += w * sm[2][c2];
    }
    out[768 + b * 128 + c] = t2[0] * z[0] + t2[1] * z[1] + t2[2] * z[2];
    float klc = 0.f;
#pragma unroll
    for (int v = 0; v < 3; ++v)
      klc += LOGK - 0.5f * lv + (sd * sd + mu[v] * mu[v]) * 8.0f - 0.5f;
    klred[c] += klc;
  }
  __syncthreads();
  if (c == 0) {
    float ssum = 0.f;
    for (int i = 0; i < 128; ++i) ssum += klred[i];
    out[1024] = ssum / (float)(BB * 128 * 3);
  }
}

// ---------------------------------------------------------------------------
// Host launch
// ---------------------------------------------------------------------------
extern "C" void kernel_launch(void* const* d_in, const int* in_sizes, int n_in,
                              void* d_out, int out_size, void* d_ws, size_t ws_size,
                              hipStream_t stream) {
  // --- input mapping (two plausible flattenings) ---
  const float *p, *conv_pos_w, *conv_pos_d, *fc_pos_w, *actvn_c_d, *fc_c_w;
  const float *fc_inv_w, *fc_mu_w, *fc_var_w, *fc_var_b, *fc_inv_2_w, *eps;
  const float *bd0[5], *bw0[5], *bd1[5], *bw1[5], *bws[5];
  if (n_in > 0 && in_sizes[0] == 768) {
    // sorted pytree order: eps, p, actvn_c_d, blk0..4(d0,d1,w0,w1,ws), conv_pos_d,
    // conv_pos_w, fc_c_w, fc_inv_2_w, fc_inv_w, fc_mu_w, fc_pos_w, fc_var_b, fc_var_w
    eps = (const float*)d_in[0];  p = (const float*)d_in[1];
    actvn_c_d = (const float*)d_in[2];
    for (int i = 0; i < 5; ++i) {
      bd0[i] = (const float*)d_in[3 + i * 5 + 0];
      bd1[i] = (const float*)d_in[3 + i * 5 + 1];
      bw0[i] = (const float*)d_in[3 + i * 5 + 2];
      bw1[i] = (const float*)d_in[3 + i * 5 + 3];
      bws[i] = (const float*)d_in[3 + i * 5 + 4];
    }
    conv_pos_d = (const float*)d_in[28]; conv_pos_w = (const float*)d_in[29];
    fc_c_w = (const float*)d_in[30]; fc_inv_2_w = (const float*)d_in[31];
    fc_inv_w = (const float*)d_in[32]; fc_mu_w = (const float*)d_in[33];
    fc_pos_w = (const float*)d_in[34]; fc_var_b = (const float*)d_in[35];
    fc_var_w = (const float*)d_in[36];
  } else {
    // insertion order
    p = (const float*)d_in[0];
    conv_pos_w = (const float*)d_in[1]; conv_pos_d = (const float*)d_in[2];
    fc_pos_w = (const float*)d_in[3];   actvn_c_d = (const float*)d_in[4];
    fc_c_w = (const float*)d_in[5];     fc_inv_w = (const float*)d_in[6];
    fc_mu_w = (const float*)d_in[7];    fc_var_w = (const float*)d_in[8];
    fc_var_b = (const float*)d_in[9];   fc_inv_2_w = (const float*)d_in[10];
    for (int i = 0; i < 5; ++i) {
      bd0[i] = (const float*)d_in[11 + i * 5 + 0];
      bw0[i] = (const float*)d_in[11 + i * 5 + 1];
      bd1[i] = (const float*)d_in[11 + i * 5 + 2];
      bw1[i] = (const float*)d_in[11 + i * 5 + 3];
      bws[i] = (const float*)d_in[11 + i * 5 + 4];
    }
    eps = (const float*)d_in[36];
  }

  // --- workspace carve-up ---
  char* ws = (char*)d_ws;
  size_t cur = 0;
  auto carve = [&](size_t bytes) -> char* {
    char* r = ws + cur;
    cur = (cur + bytes + 255) & ~(size_t)255;
    return r;
  };
  const size_t M = MTOK;
  unsigned short* WBF  = (unsigned short*)carve(851968ull * 2);     // bf16 weights
  float*          XT   = (float*)carve((size_t)BB * NPTS * 3 * 4);
  int*            IDX  = (int*)carve((size_t)BB * NPTS * KNN * 4);
  unsigned short* ACT0 = (unsigned short*)carve(M * 128 * 2);
  unsigned short* BUFA = (unsigned short*)carve(M * 256 * 2);
  unsigned short* BUFB = (unsigned short*)carve(M * 256 * 2);
  unsigned short* BUFC = (unsigned short*)carve(M * 256 * 2);
  float*          BUFF = (float*)carve(M * 128 * 4);
  float*          POOL = (float*)carve((size_t)BB * 3 * 128 * 4);
  float*          NETM = (float*)carve((size_t)BB * 3 * 128 * 4);
  (void)ws_size;

  // bf16 weight region offsets (elements)
  const size_t OFF_FCPOS = 0;
  auto blk_off = [](int i) -> size_t { return 32768 + (size_t)i * 163840; };
  // within blk: d0 +0 (65536), w0 +65536 (32768), d1 +98304 (16384),
  //             w1 +114688 (16384), ws +131072 (32768)

  auto cvt = [&](const float* src, unsigned short* dst, int n) {
    f32_to_bf16_kernel<<<(n + 255) / 256, 256, 0, stream>>>(src, dst, n);
  };
  cvt(fc_pos_w, WBF + OFF_FCPOS, 256 * 128);
  for (int i = 0; i < 5; ++i) {
    size_t o = blk_off(i);
    cvt(bd0[i], WBF + o + 0,      256 * 256);
    cvt(bw0[i], WBF + o + 65536,  128 * 256);
    cvt(bd1[i], WBF + o + 98304,  128 * 128);
    cvt(bw1[i], WBF + o + 114688, 128 * 128);
    cvt(bws[i], WBF + o + 131072, 128 * 256);
  }

  // point pipeline
  scale_transpose_kernel<<<(BB * 3 * NPTS + 255) / 256, 256, 0, stream>>>(p, XT);
  knn_kernel<<<64, 128, 0, stream>>>(XT, IDX);
  conv_pos_kernel<<<BB * NPTS, 128, 0, stream>>>(XT, IDX, conv_pos_w, conv_pos_d, ACT0);

  // fc_pos: act0 (Mx128) x fc_pos_w (256x128) -> BUFA bf16 (Mx256)
  gemm_wmma_kernel<2, 128><<<dim3(192, 4), 256, 0, stream>>>(
      ACT0, WBF + OFF_FCPOS, BUFA, 256);

  const int nlk256 = BB * NPTS * 256;
  const int nlk128 = BB * NPTS * 128;
  for (int i = 0; i < 5; ++i) {
    size_t o = blk_off(i);
    const unsigned short* Wd0 = WBF + o;
    const unsigned short* Ww0 = WBF + o + 65536;
    const unsigned short* Wd1 = WBF + o + 98304;
    const unsigned short* Ww1 = WBF + o + 114688;
    const unsigned short* Wws = WBF + o + 131072;
    // shortcut: BUFF = BUFA x ws^T  (f32)
    gemm_wmma_kernel<0, 256><<<dim3(192, 2), 256, 0, stream>>>(BUFA, Wws, BUFF, 128);
    // dir0: BUFB = BUFA x d0^T (bf16, Mx256)
    gemm_wmma_kernel<2, 256><<<dim3(192, 4), 256, 0, stream>>>(BUFA, Wd0, BUFB, 256);
    // leaky0 (slope 0): BUFC = leaky(BUFA, BUFB)
    vn_leaky_kernel<<<(nlk256 + 255) / 256, 256, 0, stream>>>(BUFA, BUFB, BUFC, 256, 0.0f, nlk256);
    // net: BUFB = BUFC x w0^T (bf16, Mx128)
    gemm_wmma_kernel<2, 256><<<dim3(192, 2), 256, 0, stream>>>(BUFC, Ww0, BUFB, 128);
    // dir1: BUFC = BUFB x d1^T (bf16, Mx128)
    gemm_wmma_kernel<2, 128><<<dim3(192, 2), 256, 0, stream>>>(BUFB, Wd1, BUFC, 128);
    // leaky1: BUFA = leaky(BUFB, BUFC)  (bf16, Mx128; BUFA free now)
    vn_leaky_kernel<<<(nlk128 + 255) / 256, 256, 0, stream>>>(BUFB, BUFC, BUFA, 128, 0.0f, nlk128);
    // dx accumulate: BUFF += BUFA x w1^T
    gemm_wmma_kernel<1, 128><<<dim3(192, 2), 256, 0, stream>>>(BUFA, Ww1, BUFF, 128);
    if (i < 4) {
      pool_mean_kernel<<<(BB * 3 * 128 + 255) / 256, 256, 0, stream>>>(BUFF, POOL, 128);
      concat_kernel<<<(unsigned)((M * 256 + 255) / 256), 256, 0, stream>>>(BUFF, POOL, BUFA);
    } else {
      pool_mean_kernel<<<(BB * 3 * 128 + 255) / 256, 256, 0, stream>>>(BUFF, NETM, 128);
    }
  }

  final_head_kernel<<<1, 128, 0, stream>>>(NETM, actvn_c_d, fc_c_w, fc_inv_w,
                                           fc_mu_w, fc_var_w, fc_var_b,
                                           fc_inv_2_w, eps, (float*)d_out);
  (void)out_size; (void)n_in;
}
